// StandardAttention_13932873908364
// MI455X (gfx1250) — matrix operand order repaired and checked by reading, not compiled, not run on hardware
//
#include <hip/hip_runtime.h>
#include <hip/hip_bf16.h>

typedef __bf16 bh16;
typedef __attribute__((ext_vector_type(16))) __bf16 v16bf;
typedef __attribute__((ext_vector_type(8)))  float  v8f;

constexpr int NB   = 2;      // batch
constexpr int SEQ  = 2048;
constexpr int HID  = 2048;
constexpr int NHq  = 16;
constexpr int NKVh = 4;
constexpr int HDm  = 128;

// Detect TDM builtin (device pass only; host pass falls back harmlessly)
#if defined(__has_builtin)
#if __has_builtin(__builtin_amdgcn_tensor_load_to_lds) && __has_builtin(__builtin_amdgcn_s_wait_tensorcnt)
#define USE_TDM 1
#endif
#endif

#ifdef USE_TDM
typedef unsigned tdm_u32x4 __attribute__((ext_vector_type(4)));
typedef int      tdm_i32x8 __attribute__((ext_vector_type(8)));
typedef int      tdm_i32x4 __attribute__((ext_vector_type(4)));

// 2D tile load via Tensor Data Mover.  D# per CDNA5 ISA ch.8:
//  group0: count=1 | lds_addr | global_addr[56:0] | type=2
//  group1: data_size=2B, pad_enable, pad_interval/amount codes,
//          tensor_dim0/1, tile_dim0/1, tensor_dim0_stride
__device__ __forceinline__ void tdm_load_2d(unsigned lds_off, const void* gptr,
                                            unsigned tile_x, unsigned tile_y,
                                            unsigned stride_x,
                                            unsigned pad_int_code, unsigned pad_amt_code) {
  unsigned long long ga = (unsigned long long)gptr;
  tdm_u32x4 g0;
  g0[0] = 1u;                                              // count=1, user desc
  g0[1] = lds_off;                                         // LDS byte address
  g0[2] = (unsigned)(ga & 0xffffffffu);                    // global_addr[31:0]
  g0[3] = (unsigned)((ga >> 32) & 0x01ffffffu) | (2u << 30); // addr[56:32] | type=2
  tdm_i32x8 g1;
  g1[0] = (int)((1u << 16)                                 // data_size: 2 bytes
              | (1u << 20)                                 // pad_enable
              | (pad_int_code << 22)
              | (pad_amt_code << 25));
  g1[1] = (int)((tile_x & 0xffffu) << 16);                 // tensor_dim0[15:0]
  g1[2] = (int)(((tile_x >> 16) & 0xffffu) | ((tile_y & 0xffffu) << 16)); // t0 hi | tensor_dim1 lo
  g1[3] = (int)(((tile_y >> 16) & 0xffffu) | ((tile_x & 0xffffu) << 16)); // t1 hi | tile_dim0
  g1[4] = (int)(tile_y & 0xffffu);                         // tile_dim1 (tile_dim2=0)
  g1[5] = (int)stride_x;                                   // tensor_dim0_stride[31:0]
  g1[6] = 0;
  g1[7] = 0;
  tdm_i32x4 z4 = {0, 0, 0, 0};
#if __clang_major__ >= 23
  tdm_i32x8 z8 = {0, 0, 0, 0, 0, 0, 0, 0};
  __builtin_amdgcn_tensor_load_to_lds(g0, g1, z4, z4, z8, 0);
#else
  __builtin_amdgcn_tensor_load_to_lds(g0, g1, z4, z4, 0);
#endif
}
#endif

// ---------------------------------------------------------------------------
// fp32 -> bf16 elementwise convert
// ---------------------------------------------------------------------------
__global__ void cvt_bf16_kernel(const float* __restrict__ in, bh16* __restrict__ out, int n) {
  int i = blockIdx.x * blockDim.x + threadIdx.x;
  if (i < n) out[i] = (bh16)in[i];
}

// ---------------------------------------------------------------------------
// WMMA fragment loaders (CDNA5 16x16x32 bf16 layouts, wave32)
// ---------------------------------------------------------------------------
__device__ __forceinline__ v16bf load_frag_a(const bh16* p, int stride, int m, int hf) {
  const bh16* r = p + m * stride;
  v16bf f;
#pragma unroll
  for (int i = 0; i < 4; i++) { int k = 8 * hf + 2 * i;      f[2*i]   = r[k]; f[2*i+1]   = r[k+1]; }
#pragma unroll
  for (int i = 0; i < 4; i++) { int k = 16 + 8 * hf + 2 * i; f[8+2*i] = r[k]; f[8+2*i+1] = r[k+1]; }
  return f;
}
__device__ __forceinline__ v16bf load_frag_b(const bh16* p, int stride, int n, int hf) {
  const bh16* r = p + n * stride;
  v16bf f;
#pragma unroll
  for (int i = 0; i < 4; i++) { int k = 8 * hf + 2 * i;      f[2*i]   = r[k]; f[2*i+1]   = r[k+1]; }
#pragma unroll
  for (int i = 0; i < 4; i++) { int k = 16 + 8 * hf + 2 * i; f[8+2*i] = r[k]; f[8+2*i+1] = r[k+1]; }
  return f;
}

// ---------------------------------------------------------------------------
// GEMM: C[M,N] (f32) = A[M,K](bf16) * W[N,K](bf16)^T
// ---------------------------------------------------------------------------
constexpr int BM = 128, BN = 128, BK = 64, LDT = 72;

__global__ __launch_bounds__(256) void gemm_bf16_kernel(
    const bh16* __restrict__ A, const bh16* __restrict__ W,
    float* __restrict__ C, int M, int N, int K) {
  __shared__ __align__(16) bh16 As[BM * LDT];
  __shared__ __align__(16) bh16 Ws[BN * LDT];
  const int tid = threadIdx.x;
  const int lane = tid & 31, wid = tid >> 5;
  const int hf = lane >> 4, ln = lane & 15;
  const int n0 = blockIdx.x * BN, m0 = blockIdx.y * BM;
  const int wm = wid & 1, wn = wid >> 1;

  v8f acc[4][2];
#pragma unroll
  for (int i = 0; i < 4; i++)
#pragma unroll
    for (int j = 0; j < 2; j++)
#pragma unroll
      for (int e = 0; e < 8; e++) acc[i][j][e] = 0.f;

  for (int kt = 0; kt < K; kt += BK) {
#pragma unroll
    for (int it = 0; it < 4; it++) {
      int q   = tid + it * 256;
      int row = q >> 3;
      int c8  = (q & 7) * 8;
      *(uint4*)&As[row * LDT + c8] = *(const uint4*)&A[(size_t)(m0 + row) * K + kt + c8];
      *(uint4*)&Ws[row * LDT + c8] = *(const uint4*)&W[(size_t)(n0 + row) * K + kt + c8];
    }
    if (kt + BK < K) {
      __builtin_prefetch(&A[(size_t)(m0 + (tid >> 1)) * K + kt + BK], 0, 1);
      __builtin_prefetch(&W[(size_t)(n0 + (tid >> 1)) * K + kt + BK], 0, 1);
    }
    __syncthreads();
#pragma unroll
    for (int ks = 0; ks < 2; ks++) {
      v16bf af[4], wf[2];
#pragma unroll
      for (int i = 0; i < 4; i++)
        af[i] = load_frag_a(&As[(wm * 64 + i * 16) * LDT + ks * 32], LDT, ln, hf);
#pragma unroll
      for (int j = 0; j < 2; j++)
        wf[j] = load_frag_b(&Ws[(wn * 32 + j * 16) * LDT + ks * 32], LDT, ln, hf);
#pragma unroll
      for (int i = 0; i < 4; i++)
#pragma unroll
        for (int j = 0; j < 2; j++)
          acc[i][j] = __builtin_amdgcn_wmma_f32_16x16x32_bf16(
              false, af[i], false, wf[j], (short)0, acc[i][j], false, false);
    }
    __syncthreads();
  }
#pragma unroll
  for (int i = 0; i < 4; i++)
#pragma unroll
    for (int j = 0; j < 2; j++) {
      int col = n0 + wn * 32 + j * 16 + ln;
#pragma unroll
      for (int e = 0; e < 8; e++) {
        int row = m0 + wm * 64 + i * 16 + e + 8 * hf;
        C[(size_t)row * N + col] = acc[i][j][e];
      }
    }
}

// ---------------------------------------------------------------------------
// RoPE + pack: X[B*S][nh*HD] f32 -> Y[B][nh][S][HD] bf16
// ---------------------------------------------------------------------------
__global__ void rope_pack_kernel(const float* __restrict__ X, bh16* __restrict__ Y, int nh) {
  int idx = blockIdx.x * blockDim.x + threadIdx.x;
  int total = NB * SEQ * nh * HDm;
  if (idx >= total) return;
  int hd = idx & (HDm - 1);
  int s  = (idx >> 7) & (SEQ - 1);
  int h  = (idx >> 18) % nh;
  int b  = idx / (262144 * nh);
  int j  = hd & 63;
  float arg = (float)s * __powf(10000.f, -(float)j * (1.f / 64.f));
  float sn, cs;
  __sincosf(arg, &sn, &cs);
  const float* row = X + (size_t)(b * SEQ + s) * (nh * HDm) + h * HDm;
  float x  = row[hd];
  float xp = (hd < 64) ? row[hd + 64] : row[hd - 64];
  float v  = (hd < 64) ? (x * cs - xp * sn) : (x * cs + xp * sn);
  Y[idx] = (bh16)v;
}

__global__ void pack_v_kernel(const float* __restrict__ X, bh16* __restrict__ Y) {
  int idx = blockIdx.x * blockDim.x + threadIdx.x;
  int total = NB * SEQ * NKVh * HDm;
  if (idx >= total) return;
  int hd = idx & (HDm - 1);
  int s  = (idx >> 7) & (SEQ - 1);
  int h  = (idx >> 18) % NKVh;
  int b  = idx / (262144 * NKVh);
  Y[idx] = (bh16)X[(size_t)(b * SEQ + s) * (NKVh * HDm) + h * HDm + hd];
}

// ---------------------------------------------------------------------------
// Flash attention (causal, GQA 4:1).  128 q rows/block, 8 waves x 16 rows.
// K tile staged via TDM (tensor_load_to_lds) when available.
// ---------------------------------------------------------------------------
constexpr int KB  = 64;
constexpr int KLD = 136;  // 128 + 8 pad (halves) == 64 + 4 dwords -> TDM pad codes 5/3
constexpr int VLD = 72;
constexpr int PLD = 72;

__global__ __launch_bounds__(256) void flash_attn_kernel(
    const bh16* __restrict__ Q, const bh16* __restrict__ Kc,
    const bh16* __restrict__ Vc, bh16* __restrict__ O) {
  __shared__ __align__(16) bh16 Klds[KB * KLD];
  __shared__ __align__(16) bh16 Vlds[HDm * VLD];
  __shared__ __align__(16) bh16 Plds[8 * 16 * PLD];
  const int tid = threadIdx.x, lane = tid & 31, wid = tid >> 5;
  const int hf = lane >> 4, ln = lane & 15;
  const int qb = blockIdx.x;
  const int bh = blockIdx.y;
  const int b = bh / NHq, h = bh % NHq;
  const int q0 = qb * 128;
  const int qrow0 = q0 + wid * 16;
  const bh16* Qg = Q + ((size_t)(b * NHq + h) * SEQ + qrow0) * HDm;
  const bh16* Kg = Kc + (size_t)(b * NKVh + h / 4) * SEQ * HDm;
  const bh16* Vg = Vc + (size_t)(b * NKVh + h / 4) * SEQ * HDm;
  bh16* Pw = &Plds[wid * 16 * PLD];

  v16bf qf[4];
#pragma unroll
  for (int kf = 0; kf < 4; kf++) qf[kf] = load_frag_a(Qg + kf * 32, HDm, ln, hf);

  v8f o[8];
#pragma unroll
  for (int t = 0; t < 8; t++)
#pragma unroll
    for (int e = 0; e < 8; e++) o[t][e] = 0.f;
  float mrow[8], lrow[8];
#pragma unroll
  for (int e = 0; e < 8; e++) { mrow[e] = -1e30f; lrow[e] = 0.f; }
  const float scale = 0.08838834764831845f;
  const int nkb = 2 * (qb + 1);

  for (int kb = 0; kb < nkb; kb++) {
    const int k0 = kb * KB;
    // ---- stage K tile 64x128 ----
#ifdef USE_TDM
    if (wid == 0) {
      // tile: 128 elems/row (x), 64 rows (y); row stride 128 elems; LDS rows
      // padded by 4 dwords every 64 dwords -> KLD=136 halves.
      tdm_load_2d((unsigned)(uintptr_t)&Klds[0], Kg + (size_t)k0 * HDm,
                  HDm, KB, HDm, /*pad_interval 64dw*/5u, /*pad_amount 4dw*/3u);
    }
#else
#pragma unroll
    for (int it = 0; it < 4; it++) {
      int q   = tid + it * 256;
      int row = q >> 3;
      int c8  = (q & 7) * 8;
      *(uint4*)&Klds[row * KLD + c8] = *(const uint4*)&Kg[(size_t)(k0 + row) * HDm + c8];
    }
#endif
    // ---- stage V transposed: Vlds[hd][key] (TDM cannot transpose) ----
#pragma unroll
    for (int it = 0; it < 16; it++) {
      int e2  = tid + it * 256;        // 4096 dword ids
      int key = e2 >> 6;               // 64 dwords per key row
      int hd2 = (e2 & 63) * 2;
      union { unsigned u; bh16 hv[2]; } cv;
      cv.u = *(const unsigned*)&Vg[(size_t)(k0 + key) * HDm + hd2];
      Vlds[hd2 * VLD + key]       = cv.hv[0];
      Vlds[(hd2 + 1) * VLD + key] = cv.hv[1];
    }
#ifdef USE_TDM
    if (wid == 0) __builtin_amdgcn_s_wait_tensorcnt(0);
#endif
    __syncthreads();

    // ---- S = Q K^T ----
    v8f st[4];
#pragma unroll
    for (int t = 0; t < 4; t++)
#pragma unroll
      for (int e = 0; e < 8; e++) st[t][e] = 0.f;
#pragma unroll
    for (int t = 0; t < 4; t++)
#pragma unroll
      for (int kf = 0; kf < 4; kf++) {
        v16bf kfrag = load_frag_b(&Klds[t * 16 * KLD + kf * 32], KLD, ln, hf);
        st[t] = __builtin_amdgcn_wmma_f32_16x16x32_bf16(
            false, qf[kf], false, kfrag, (short)0, st[t], false, false);
      }
    // ---- scale + causal mask ----
#pragma unroll
    for (int t = 0; t < 4; t++) {
      int key = k0 + t * 16 + ln;
#pragma unroll
      for (int e = 0; e < 8; e++) {
        int qr = qrow0 + e + 8 * hf;
        float sv = st[t][e] * scale;
        st[t][e] = (key <= qr) ? sv : -1e9f;
      }
    }
    // ---- online softmax ----
#pragma unroll
    for (int e = 0; e < 8; e++) {
      float v = fmaxf(fmaxf(st[0][e], st[1][e]), fmaxf(st[2][e], st[3][e]));
      v = fmaxf(v, __shfl_xor(v, 1, 32));
      v = fmaxf(v, __shfl_xor(v, 2, 32));
      v = fmaxf(v, __shfl_xor(v, 4, 32));
      v = fmaxf(v, __shfl_xor(v, 8, 32));
      float mnew = fmaxf(mrow[e], v);
      float alpha = __expf(mrow[e] - mnew);
      float psum = 0.f;
#pragma unroll
      for (int t = 0; t < 4; t++) {
        float p = __expf(st[t][e] - mnew);
        st[t][e] = p;
        psum += p;
      }
      psum += __shfl_xor(psum, 1, 32);
      psum += __shfl_xor(psum, 2, 32);
      psum += __shfl_xor(psum, 4, 32);
      psum += __shfl_xor(psum, 8, 32);
      lrow[e] = lrow[e] * alpha + psum;
      mrow[e] = mnew;
#pragma unroll
      for (int t = 0; t < 8; t++) o[t][e] *= alpha;
    }
    // ---- P -> per-wave LDS (D-layout -> A-layout) ----
#pragma unroll
    for (int t = 0; t < 4; t++)
#pragma unroll
      for (int e = 0; e < 8; e++)
        Pw[(e + 8 * hf) * PLD + t * 16 + ln] = (bh16)st[t][e];
    // ---- O += P V ----
#pragma unroll
    for (int kf = 0; kf < 2; kf++) {
      v16bf pf = load_frag_a(Pw + kf * 32, PLD, ln, hf);
#pragma unroll
      for (int t = 0; t < 8; t++) {
        v16bf vf = load_frag_b(&Vlds[t * 16 * VLD + kf * 32], VLD, ln, hf);
        o[t] = __builtin_amdgcn_wmma_f32_16x16x32_bf16(
            false, pf, false, vf, (short)0, o[t], false, false);
      }
    }
    __syncthreads();
  }
  bh16* Og = O + (size_t)(b * SEQ + qrow0) * (NHq * HDm) + h * HDm;
#pragma unroll
  for (int e = 0; e < 8; e++) {
    float inv = 1.f / lrow[e];
#pragma unroll
    for (int t = 0; t < 8; t++)
      Og[(size_t)(e + 8 * hf) * (NHq * HDm) + t * 16 + ln] = (bh16)(o[t][e] * inv);
  }
}

// ---------------------------------------------------------------------------
extern "C" void kernel_launch(void* const* d_in, const int* in_sizes, int n_in,
                              void* d_out, int out_size, void* d_ws, size_t ws_size,
                              hipStream_t stream) {
  (void)in_sizes; (void)n_in; (void)out_size; (void)ws_size;
  const float* X  = (const float*)d_in[0];
  const float* Wq = (const float*)d_in[3];
  const float* Wk = (const float*)d_in[4];
  const float* Wv = (const float*)d_in[5];
  const float* Wo = (const float*)d_in[6];
  float* out = (float*)d_out;

  const int M = NB * SEQ;
  const int KVD = NKVh * HDm;
  char* ws = (char*)d_ws;
  size_t off = 0;
  auto carve = [&](size_t bytes) { char* p = ws + off; off += (bytes + 255) & ~(size_t)255; return p; };
  bh16* Xbf = (bh16*)carve((size_t)M * HID * 2);
  bh16* Wqb = (bh16*)carve((size_t)HID * HID * 2);
  bh16* Wkb = (bh16*)carve((size_t)KVD * HID * 2);
  bh16* Wvb = (bh16*)carve((size_t)KVD * HID * 2);
  bh16* Wob = (bh16*)carve((size_t)HID * HID * 2);
  float* Qf = (float*)carve((size_t)M * HID * 4);
  float* Kf = (float*)carve((size_t)M * KVD * 4);
  float* Vf = (float*)carve((size_t)M * KVD * 4);
  bh16* Qp = (bh16*)carve((size_t)M * HID * 2);
  bh16* Kp = (bh16*)carve((size_t)M * KVD * 2);
  bh16* Vp = (bh16*)carve((size_t)M * KVD * 2);
  bh16* AO = (bh16*)carve((size_t)M * HID * 2);

  auto g1 = [](int n) { return dim3((n + 255) / 256); };
  cvt_bf16_kernel<<<g1(M * HID), 256, 0, stream>>>(X, Xbf, M * HID);
  cvt_bf16_kernel<<<g1(HID * HID), 256, 0, stream>>>(Wq, Wqb, HID * HID);
  cvt_bf16_kernel<<<g1(KVD * HID), 256, 0, stream>>>(Wk, Wkb, KVD * HID);
  cvt_bf16_kernel<<<g1(KVD * HID), 256, 0, stream>>>(Wv, Wvb, KVD * HID);
  cvt_bf16_kernel<<<g1(HID * HID), 256, 0, stream>>>(Wo, Wob, HID * HID);

  gemm_bf16_kernel<<<dim3(HID / BN, M / BM), 256, 0, stream>>>(Xbf, Wqb, Qf, M, HID, HID);
  gemm_bf16_kernel<<<dim3(KVD / BN, M / BM), 256, 0, stream>>>(Xbf, Wkb, Kf, M, KVD, HID);
  gemm_bf16_kernel<<<dim3(KVD / BN, M / BM), 256, 0, stream>>>(Xbf, Wvb, Vf, M, KVD, HID);

  rope_pack_kernel<<<g1(M * HID), 256, 0, stream>>>(Qf, Qp, NHq);
  rope_pack_kernel<<<g1(M * KVD), 256, 0, stream>>>(Kf, Kp, NKVh);
  pack_v_kernel<<<g1(M * KVD), 256, 0, stream>>>(Vf, Vp);

  flash_attn_kernel<<<dim3(SEQ / 128, NB * NHq), 256, 0, stream>>>(Qp, Kp, Vp, AO);

  gemm_bf16_kernel<<<dim3(HID / BN, M / BM), 256, 0, stream>>>(AO, Wob, out, M, HID, HID);
}
